// Postal_temporal_GraphSage_att_78099685310581
// MI455X (gfx1250) — matrix operand order repaired and checked by reading, not compiled, not run on hardware
//
#include <hip/hip_runtime.h>

#define DEVFN __device__ __forceinline__

typedef __attribute__((ext_vector_type(16))) __bf16 v16bf;
typedef __attribute__((ext_vector_type(8)))  float  v8f;

union BFrag { v16bf v; uint4 q[2]; };

// ---------------- constants (match reference) ----------------
static constexpr int T_ = 12, IN_ = 12, H_ = 128;

// per-wave LDS scratch sizes (bytes)
static constexpr int PW_ATTN = 16*136*2*3 + 128*40*2 + 16*16*4 + 16*40*2; // 25600
static constexpr int PW_FC   = 16*168*2 + 16*264*2 + 16*132*4;            // 22272
static constexpr int PW_C1   = 8192;                                      // >= max(stage 2560, out 16x128 f32)
static constexpr int PW_C2   = 8704;                                      // >= max(stage 8704, out 8192)

// ---------------- helpers ----------------
DEVFN unsigned short f2bf(float f) {
  unsigned int u = __float_as_uint(f);
  unsigned int r = u + 0x7FFFu + ((u >> 16) & 1u);   // RNE truncate to bf16
  return (unsigned short)(r >> 16);
}

DEVFN v8f zero8() {
  v8f c;
#pragma unroll
  for (int i = 0; i < 8; ++i) c[i] = 0.0f;
  return c;
}

DEVFN v8f wmma_bf16(v16bf a, v16bf b, v8f c) {
  // D = A(16x32 bf16) * B(32x16 bf16) + C(16x16 f32)
  return __builtin_amdgcn_wmma_f32_16x16x32_bf16(false, a, false, b, (short)0, c, false, false);
}

// A fragment: A[m][k], m = lane&15; k layout per ISA 7.12.2 (two b128 LDS reads)
DEVFN v16bf load_a(const unsigned short* base, int lda, int kstep, int lane) {
  int m = lane & 15, kh = lane >> 4;
  const unsigned short* row = base + m * lda + kstep * 32 + kh * 8;
  BFrag f;
  f.q[0] = *(const uint4*)(row);
  f.q[1] = *(const uint4*)(row + 16);
  return f.v;
}

// B fragment: B[k][n] = Ws[row0+n][k]  (weight rows row-major, k layout matches load_a)
DEVFN v16bf load_b(const unsigned short* base, int lda, int row0, int kstep, int lane) {
  int nn = lane & 15, kh = lane >> 4;
  const unsigned short* row = base + (row0 + nn) * lda + kstep * 32 + kh * 8;
  BFrag f;
  f.q[0] = *(const uint4*)(row);
  f.q[1] = *(const uint4*)(row + 16);
  return f.v;
}

// ---------------- graph kernels ----------------
__global__ void k_degree(const int* __restrict__ ei, int E, float* __restrict__ deg) {
  int e = blockIdx.x * blockDim.x + threadIdx.x;
  if (e < E) atomicAdd(&deg[ei[E + e]], 1.0f);
}

__global__ void k_deginv(const float* __restrict__ deg, float* __restrict__ dinv, int N) {
  int n = blockIdx.x * blockDim.x + threadIdx.x;
  if (n < N) dinv[n] = 1.0f / fmaxf(deg[n], 1.0f);
}

__global__ void k_scatter12(const int* __restrict__ ei, int E, int N,
                            const float* __restrict__ x, float* __restrict__ agg) {
  int e = blockIdx.x * blockDim.x + threadIdx.x;
  int t = blockIdx.y;
  if (e >= E) return;
  int s = ei[e], d = ei[E + e];
  const float* xr = x + ((size_t)t * N + s) * 12;
  float* ar = agg + ((size_t)t * N + d) * 12;
  float4 a = *(const float4*)(xr), b = *(const float4*)(xr + 4), c = *(const float4*)(xr + 8);
  atomicAdd(ar + 0, a.x);  atomicAdd(ar + 1, a.y);  atomicAdd(ar + 2, a.z);  atomicAdd(ar + 3, a.w);
  atomicAdd(ar + 4, b.x);  atomicAdd(ar + 5, b.y);  atomicAdd(ar + 6, b.z);  atomicAdd(ar + 7, b.w);
  atomicAdd(ar + 8, c.x);  atomicAdd(ar + 9, c.y);  atomicAdd(ar + 10, c.z); atomicAdd(ar + 11, c.w);
}

__global__ void k_scatter128(const int* __restrict__ ei, int E, int N,
                             const float* __restrict__ hin, float* __restrict__ agg) {
  long long tid = (long long)blockIdx.x * blockDim.x + threadIdx.x;
  if (tid >= (long long)E * 8) return;
  int e = (int)(tid >> 3), ch = (int)(tid & 7);
  int t = blockIdx.y;
  int s = ei[e], d = ei[E + e];
  const float* xr = hin + ((size_t)t * N + s) * 128 + ch * 16;
  float* ar = agg + ((size_t)t * N + d) * 128 + ch * 16;
#pragma unroll
  for (int i = 0; i < 4; ++i) {
    float4 v = *(const float4*)(xr + i * 4);
    atomicAdd(ar + i * 4 + 0, v.x);
    atomicAdd(ar + i * 4 + 1, v.y);
    atomicAdd(ar + i * 4 + 2, v.z);
    atomicAdd(ar + i * 4 + 3, v.w);
  }
}

// ---------------- conv1: relu(mean@Wl^T + x@Wr^T + b), K=12 (pad 32), persistent ----------------
__global__ void __launch_bounds__(256) k_conv1(const float* __restrict__ x,
                                               const float* __restrict__ agg,
                                               const float* __restrict__ dinv,
                                               const float* __restrict__ Wl,
                                               const float* __restrict__ Wr,
                                               const float* __restrict__ bias,
                                               float* __restrict__ out, int N, int ntiles) {
  extern __shared__ __align__(16) char smem[];
  unsigned short* sWl = (unsigned short*)smem;      // 128x40
  unsigned short* sWr = sWl + 128 * 40;             // 128x40
  float* sB = (float*)(sWr + 128 * 40);             // 128
  char* pw = (char*)(sB + 128);
  int wave = threadIdx.x >> 5, lane = threadIdx.x & 31;
  unsigned short* xs = (unsigned short*)(pw + wave * PW_C1);
  unsigned short* ms = xs + 16 * 40;
  float* outs = (float*)xs;                         // 16x128 f32, reused after frag loads

  for (int idx = threadIdx.x; idx < 128 * 40; idx += 256) {
    int r = idx / 40, c = idx - r * 40;
    sWl[idx] = (c < 12) ? f2bf(Wl[r * 12 + c]) : (unsigned short)0;
    sWr[idx] = (c < 12) ? f2bf(Wr[r * 12 + c]) : (unsigned short)0;
  }
  for (int idx = threadIdx.x; idx < 128; idx += 256) sB[idx] = bias[idx];
  __syncthreads();

  int ntasks = ntiles * T_;
  int nwaves = gridDim.x * 8;
  int col = lane & 15, kh = lane >> 4;

  for (int task = blockIdx.x * 8 + wave; task < ntasks; task += nwaves) {
    int t = task / ntiles, tile = task - t * ntiles;
    int n0 = tile * 16;

    { // prefetch next task's rows (global_prefetch_b8)
      int task2 = task + nwaves;
      if (task2 < ntasks && lane < 16) {
        int t2 = task2 / ntiles, n02 = (task2 - t2 * ntiles) * 16;
        if (n02 + lane < N) {
          __builtin_prefetch(x + ((size_t)t2 * N + n02 + lane) * 12, 0, 1);
          __builtin_prefetch(agg + ((size_t)t2 * N + n02 + lane) * 12, 0, 1);
        }
      }
    }

    if (lane < 16) { // stage 16 node rows (K padded 12->32 with zeros)
      int r = lane;
      bool ok = (n0 + r) < N;
      const float* xr = x + ((size_t)t * N + n0 + r) * 12;
      const float* ar = agg + ((size_t)t * N + n0 + r) * 12;
      float di = ok ? dinv[n0 + r] : 0.0f;
#pragma unroll
      for (int c = 0; c < 40; ++c) {
        xs[r * 40 + c] = (ok && c < 12) ? f2bf(xr[c]) : (unsigned short)0;
        ms[r * 40 + c] = (ok && c < 12) ? f2bf(ar[c] * di) : (unsigned short)0;
      }
    }
    v16bf ax = load_a(xs, 40, 0, lane);
    v16bf am = load_a(ms, 40, 0, lane);
#pragma unroll
    for (int ot = 0; ot < 8; ++ot) {
      v8f c = zero8();
      c = wmma_bf16(am, load_b(sWl, 40, ot * 16, 0, lane), c);
      c = wmma_bf16(ax, load_b(sWr, 40, ot * 16, 0, lane), c);
      float bb = sB[ot * 16 + col];
#pragma unroll
      for (int r = 0; r < 8; ++r) {
        float v = c[r] + bb;
        outs[(r + 8 * kh) * 128 + ot * 16 + col] = v > 0.f ? v : 0.f;
      }
    }
    // coalesced row writeback: 512B contiguous per row per wave
#pragma unroll
    for (int i = 0; i < 16; ++i) {
      if (n0 + i < N) {
        float4 v = *(const float4*)(outs + i * 128 + lane * 4);
        *(float4*)(out + ((size_t)t * N + n0 + i) * 128 + lane * 4) = v;
      }
    }
  }
}

// ---------------- fused per-node temporal attention (one wave per node, persistent) ----------------
__global__ void __launch_bounds__(256) k_attn(const float* __restrict__ in,
                                              float* __restrict__ outTN,
                                              float* __restrict__ outSum,
                                              const float* __restrict__ Wq, const float* __restrict__ bq,
                                              const float* __restrict__ Wk, const float* __restrict__ bk,
                                              const float* __restrict__ Wv, const float* __restrict__ bv,
                                              int N, int sumMode) {
  extern __shared__ __align__(16) char smem[];
  unsigned short* sWq = (unsigned short*)smem;   // 128x136 bf16 each
  unsigned short* sWk = sWq + 128 * 136;
  unsigned short* sWv = sWk + 128 * 136;
  float* sbq = (float*)(sWv + 128 * 136);
  float* sbk = sbq + 128;
  float* sbv = sbk + 128;
  char* pw = (char*)(sbv + 128);

  int wave = threadIdx.x >> 5, lane = threadIdx.x & 31;
  char* my = pw + wave * PW_ATTN;
  unsigned short* Xs  = (unsigned short*)my;     // 16x136
  unsigned short* qs  = Xs + 16 * 136;           // 16x136
  unsigned short* kks = qs + 16 * 136;           // 16x136
  unsigned short* vts = kks + 16 * 136;          // 128x40 (v transposed)
  float* ss = (float*)(vts + 128 * 40);          // 16x16 scores
  unsigned short* wsm = (unsigned short*)(ss + 256); // 16x40 softmax weights bf16
  float* po = (float*)Xs;                        // output staging (reuses Xs/qs area)

  for (int idx = threadIdx.x; idx < 128 * 136; idx += 256) {
    int r = idx / 136, c = idx - r * 136;
    bool v = c < 128;
    sWq[idx] = v ? f2bf(Wq[r * 128 + c]) : (unsigned short)0;
    sWk[idx] = v ? f2bf(Wk[r * 128 + c]) : (unsigned short)0;
    sWv[idx] = v ? f2bf(Wv[r * 128 + c]) : (unsigned short)0;
  }
  for (int idx = threadIdx.x; idx < 128; idx += 256) {
    sbq[idx] = bq[idx]; sbk[idx] = bk[idx]; sbv[idx] = bv[idx];
  }
  { // zero all per-wave scratch once (establishes zero K-pads)
    unsigned int* z = (unsigned int*)my;
    for (int i = lane; i < PW_ATTN / 4; i += 32) z[i] = 0u;
  }
  __syncthreads();

  int nwaves = gridDim.x * 8;
  int col = lane & 15, kh = lane >> 4;
  const float qscale = 0.08838834764831845f; // 1/sqrt(128)

  for (int n = blockIdx.x * 8 + wave; n < N; n += nwaves) {
    { // prefetch next node's feature rows
      int n2 = n + nwaves;
      int r = lane >> 1, half = lane & 1;
      if (n2 < N && r < 12)
        __builtin_prefetch(in + ((size_t)r * N + n2) * 128 + half * 64, 0, 1);
    }
    { // re-zero Xs rows 12..15 (clobbered by output staging; WMMA propagates NaN through 0*NaN)
      unsigned int* z = (unsigned int*)(Xs + 12 * 136);
      for (int i = lane; i < 272; i += 32) z[i] = 0u;
    }
    { // stage X[t=0..11][0..127] -> bf16
      int r = lane >> 1, half = lane & 1;
      if (r < 12) {
        const float* src = in + ((size_t)r * N + n) * 128;
#pragma unroll
        for (int i = 0; i < 16; ++i) {
          int c = half * 64 + i * 4;
          float4 v = *(const float4*)(src + c);
          Xs[r * 136 + c + 0] = f2bf(v.x);
          Xs[r * 136 + c + 1] = f2bf(v.y);
          Xs[r * 136 + c + 2] = f2bf(v.z);
          Xs[r * 136 + c + 3] = f2bf(v.w);
        }
      }
    }

    v16bf ax[4];
#pragma unroll
    for (int ks = 0; ks < 4; ++ks) ax[ks] = load_a(Xs, 136, ks, lane);

    // q, k, v linears: 96 WMMAs per node
#pragma unroll
    for (int w = 0; w < 3; ++w) {
      const unsigned short* Wm = (w == 0) ? sWq : (w == 1) ? sWk : sWv;
      const float* bm = (w == 0) ? sbq : (w == 1) ? sbk : sbv;
#pragma unroll
      for (int ot = 0; ot < 8; ++ot) {
        v8f c = zero8();
#pragma unroll
        for (int ks = 0; ks < 4; ++ks)
          c = wmma_bf16(ax[ks], load_b(Wm, 136, ot * 16, ks, lane), c);
        float bb = bm[ot * 16 + col];
#pragma unroll
        for (int r = 0; r < 8; ++r) {
          int m = r + 8 * kh;
          float v = c[r] + bb;
          if (w == 0)      qs[m * 136 + ot * 16 + col] = f2bf(v * qscale);
          else if (w == 1) kks[m * 136 + ot * 16 + col] = f2bf(v);
          else             vts[(ot * 16 + col) * 40 + m] = f2bf(v);  // transposed for B-side reuse
        }
      }
    }

    { // scores[t][s] = q[t] . k[s]  (K=128)
      v16bf aq[4];
#pragma unroll
      for (int ks = 0; ks < 4; ++ks) aq[ks] = load_a(qs, 136, ks, lane);
      v8f c = zero8();
#pragma unroll
      for (int ks = 0; ks < 4; ++ks)
        c = wmma_bf16(aq[ks], load_b(kks, 136, 0, ks, lane), c);
#pragma unroll
      for (int r = 0; r < 8; ++r) ss[(r + 8 * kh) * 16 + col] = c[r];
    }

    // masked softmax over valid 12x12 (wave-local LDS, in order)
    if (lane < 12) {
      float mx = -1e30f;
#pragma unroll
      for (int cc = 0; cc < 12; ++cc) mx = fmaxf(mx, ss[lane * 16 + cc]);
      float ex[12]; float sum = 0.f;
#pragma unroll
      for (int cc = 0; cc < 12; ++cc) { ex[cc] = __expf(ss[lane * 16 + cc] - mx); sum += ex[cc]; }
      float inv = 1.0f / sum;
#pragma unroll
      for (int cc = 0; cc < 12; ++cc) wsm[lane * 40 + cc] = f2bf(ex[cc] * inv);
    }

    // out = w @ v
    v16bf aw = load_a(wsm, 40, 0, lane);
    if (!sumMode) {
#pragma unroll
      for (int ot = 0; ot < 8; ++ot) {
        v8f c = zero8();
        c = wmma_bf16(aw, load_b(vts, 40, ot * 16, 0, lane), c);
#pragma unroll
        for (int r = 0; r < 8; ++r)
          po[(r + 8 * kh) * 128 + ot * 16 + col] = c[r];
      }
      // coalesced 512B-per-row writeback of valid rows t=0..11
#pragma unroll
      for (int i = 0; i < 12; ++i) {
        float4 v = *(const float4*)(po + i * 128 + lane * 4);
        *(float4*)(outTN + ((size_t)i * N + n) * 128 + lane * 4) = v;
      }
    } else {
#pragma unroll
      for (int ot = 0; ot < 8; ++ot) {
        v8f c = zero8();
        c = wmma_bf16(aw, load_b(vts, 40, ot * 16, 0, lane), c);
        float s = 0.f;
#pragma unroll
        for (int r = 0; r < 8; ++r) s += c[r];   // rows >=12 contribute exactly zero
        s += __shfl_xor(s, 16, 32);
        if (lane < 16) po[ot * 16 + lane] = s;
      }
      float4 v = *(const float4*)(po + lane * 4);
      *(float4*)(outSum + (size_t)n * 128 + lane * 4) = v;
    }
  }
}

// ---------------- conv2: relu(mean@Wl^T + h@Wr^T + b), K=128, persistent ----------------
__global__ void __launch_bounds__(256) k_conv2(const float* __restrict__ hin,
                                               const float* __restrict__ agg,
                                               const float* __restrict__ dinv,
                                               const float* __restrict__ Wl,
                                               const float* __restrict__ Wr,
                                               const float* __restrict__ bias,
                                               float* __restrict__ out, int N, int ntiles) {
  extern __shared__ __align__(16) char smem[];
  unsigned short* sWl = (unsigned short*)smem;   // 128x136
  unsigned short* sWr = sWl + 128 * 136;
  float* sB = (float*)(sWr + 128 * 136);
  char* pw = (char*)(sB + 128);
  int wave = threadIdx.x >> 5, lane = threadIdx.x & 31;
  unsigned short* hs = (unsigned short*)(pw + wave * PW_C2);
  unsigned short* ms = hs + 16 * 136;
  float* outs = (float*)hs;                      // 16x128 f32, reused after frag loads

  for (int idx = threadIdx.x; idx < 128 * 136; idx += 256) {
    int r = idx / 136, c = idx - r * 136;
    bool v = c < 128;
    sWl[idx] = v ? f2bf(Wl[r * 128 + c]) : (unsigned short)0;
    sWr[idx] = v ? f2bf(Wr[r * 128 + c]) : (unsigned short)0;
  }
  for (int idx = threadIdx.x; idx < 128; idx += 256) sB[idx] = bias[idx];
  __syncthreads();

  int ntasks = ntiles * T_;
  int nwaves = gridDim.x * 8;
  int col = lane & 15, kh = lane >> 4;

  for (int task = blockIdx.x * 8 + wave; task < ntasks; task += nwaves) {
    int t = task / ntiles, tile = task - t * ntiles;
    int n0 = tile * 16;

    { // prefetch next task's rows
      int task2 = task + nwaves;
      if (task2 < ntasks) {
        int t2 = task2 / ntiles, n02 = (task2 - t2 * ntiles) * 16;
        int r = lane >> 1, half = lane & 1;
        if (n02 + r < N) {
          __builtin_prefetch(hin + ((size_t)t2 * N + n02 + r) * 128 + half * 64, 0, 1);
          __builtin_prefetch(agg + ((size_t)t2 * N + n02 + r) * 128 + half * 64, 0, 1);
        }
      }
    }

    { // stage h and mean rows (lane pairs: 16 rows x 2 halves)
      int r = lane >> 1, half = lane & 1;
      bool ok = (n0 + r) < N;
      const float* hr = hin + ((size_t)t * N + n0 + r) * 128;
      const float* ar = agg + ((size_t)t * N + n0 + r) * 128;
      float di = ok ? dinv[n0 + r] : 0.0f;
#pragma unroll
      for (int i = 0; i < 16; ++i) {
        int c = half * 64 + i * 4;
        float4 hv = ok ? *(const float4*)(hr + c) : float4{0, 0, 0, 0};
        float4 av = ok ? *(const float4*)(ar + c) : float4{0, 0, 0, 0};
        hs[r * 136 + c + 0] = f2bf(hv.x); hs[r * 136 + c + 1] = f2bf(hv.y);
        hs[r * 136 + c + 2] = f2bf(hv.z); hs[r * 136 + c + 3] = f2bf(hv.w);
        ms[r * 136 + c + 0] = f2bf(av.x * di); ms[r * 136 + c + 1] = f2bf(av.y * di);
        ms[r * 136 + c + 2] = f2bf(av.z * di); ms[r * 136 + c + 3] = f2bf(av.w * di);
      }
    }

    v16bf ah[4], am[4];
#pragma unroll
    for (int ks = 0; ks < 4; ++ks) { ah[ks] = load_a(hs, 136, ks, lane); am[ks] = load_a(ms, 136, ks, lane); }

#pragma unroll
    for (int ot = 0; ot < 8; ++ot) {
      v8f c = zero8();
#pragma unroll
      for (int ks = 0; ks < 4; ++ks) {
        c = wmma_bf16(am[ks], load_b(sWl, 136, ot * 16, ks, lane), c);
        c = wmma_bf16(ah[ks], load_b(sWr, 136, ot * 16, ks, lane), c);
      }
      float bb = sB[ot * 16 + col];
#pragma unroll
      for (int r = 0; r < 8; ++r) {
        float v = c[r] + bb;
        outs[(r + 8 * kh) * 128 + ot * 16 + col] = v > 0.f ? v : 0.f;
      }
    }
    // coalesced row writeback
#pragma unroll
    for (int i = 0; i < 16; ++i) {
      if (n0 + i < N) {
        float4 v = *(const float4*)(outs + i * 128 + lane * 4);
        *(float4*)(out + ((size_t)t * N + n0 + i) * 128 + lane * 4) = v;
      }
    }
  }
}

// ---------------- fused 3-layer MLP head ----------------
__global__ void __launch_bounds__(128) k_fc(const float* __restrict__ femb,
                                            const int* __restrict__ nidx,
                                            const float* __restrict__ apart,
                                            const float* __restrict__ W1, const float* __restrict__ b1,
                                            const float* __restrict__ W2, const float* __restrict__ b2,
                                            const float* __restrict__ W3, const float* __restrict__ b3,
                                            float* __restrict__ out, int Bn) {
  extern __shared__ __align__(16) char smem[];
  unsigned short* sW1 = (unsigned short*)smem;   // 256x168
  unsigned short* sW2 = sW1 + 256 * 168;         // 128x264
  float* sb1 = (float*)(sW2 + 128 * 264);        // 256
  float* sb2 = sb1 + 256;                        // 128
  float* sw3 = sb2 + 128;                        // 128
  float* sb3 = sw3 + 128;                        // 4 (pad)
  char* pw = (char*)(sb3 + 4);
  int wave = threadIdx.x >> 5, lane = threadIdx.x & 31;
  char* my = pw + wave * PW_FC;
  unsigned short* in0 = (unsigned short*)my;     // 16x168
  unsigned short* h1 = in0 + 16 * 168;           // 16x264
  float* h2 = (float*)(h1 + 16 * 264);           // 16x132 f32

  for (int idx = threadIdx.x; idx < 256 * 168; idx += 128) {
    int r = idx / 168, c = idx - r * 168;
    sW1[idx] = (c < 138) ? f2bf(W1[r * 138 + c]) : (unsigned short)0;
  }
  for (int idx = threadIdx.x; idx < 128 * 264; idx += 128) {
    int r = idx / 264, c = idx - r * 264;
    sW2[idx] = (c < 256) ? f2bf(W2[r * 256 + c]) : (unsigned short)0;
  }
  for (int idx = threadIdx.x; idx < 256; idx += 128) sb1[idx] = b1[idx];
  for (int idx = threadIdx.x; idx < 128; idx += 128) { sb2[idx] = b2[idx]; sw3[idx] = W3[idx]; }
  if (threadIdx.x == 0) sb3[0] = b3[0];
  { unsigned int* z = (unsigned int*)my; for (int i = lane; i < PW_FC / 4; i += 32) z[i] = 0u; }
  __syncthreads();

  int tile = blockIdx.x * 4 + wave;
  int row0 = tile * 16;
  if (row0 >= Bn) return;

  // stage concat(femb[node_idx], apart) as bf16, pad to 160 with zeros
  for (int idx = lane; idx < 16 * 168; idx += 32) {
    int r = idx / 168, c = idx - r * 168;
    float v = 0.f;
    if (row0 + r < Bn) {
      if (c < 128)      v = femb[(size_t)nidx[row0 + r] * 128 + c];
      else if (c < 138) v = apart[(size_t)(row0 + r) * 10 + (c - 128)];
    }
    in0[idx] = f2bf(v);
  }

  int col = lane & 15, kh = lane >> 4;
  // fc1: [16,160] x [160,256], leaky 0.1
  v16bf a1[5];
#pragma unroll
  for (int ks = 0; ks < 5; ++ks) a1[ks] = load_a(in0, 168, ks, lane);
#pragma unroll
  for (int ot = 0; ot < 16; ++ot) {
    v8f c = zero8();
#pragma unroll
    for (int ks = 0; ks < 5; ++ks) c = wmma_bf16(a1[ks], load_b(sW1, 168, ot * 16, ks, lane), c);
    float bb = sb1[ot * 16 + col];
#pragma unroll
    for (int r = 0; r < 8; ++r) {
      int m = r + 8 * kh;
      float v = c[r] + bb;
      h1[m * 264 + ot * 16 + col] = f2bf(v > 0.f ? v : 0.1f * v);
    }
  }
  // fc2: [16,256] x [256,128], leaky 0.05
  v16bf a2[8];
#pragma unroll
  for (int ks = 0; ks < 8; ++ks) a2[ks] = load_a(h1, 264, ks, lane);
#pragma unroll
  for (int ot = 0; ot < 8; ++ot) {
    v8f c = zero8();
#pragma unroll
    for (int ks = 0; ks < 8; ++ks) c = wmma_bf16(a2[ks], load_b(sW2, 264, ot * 16, ks, lane), c);
    float bb = sb2[ot * 16 + col];
#pragma unroll
    for (int r = 0; r < 8; ++r) {
      int m = r + 8 * kh;
      float v = c[r] + bb;
      h2[m * 132 + ot * 16 + col] = v > 0.f ? v : 0.05f * v;
    }
  }
  // fc3: dot(h2[r], w3) + b3
  {
    int r = lane >> 1, half = lane & 1;
    float p = 0.f;
    for (int i = 0; i < 64; ++i) { int c2 = half * 64 + i; p += h2[r * 132 + c2] * sw3[c2]; }
    p += __shfl_xor(p, 1, 32);
    if (half == 0 && (row0 + r) < Bn) out[row0 + r] = p + sb3[0];
  }
}

// ---------------- host launch ----------------
extern "C" void kernel_launch(void* const* d_in, const int* in_sizes, int n_in,
                              void* d_out, int out_size, void* d_ws, size_t ws_size,
                              hipStream_t stream) {
  const float* x     = (const float*)d_in[0];
  const int*   ei    = (const int*)d_in[1];
  const int*   nidx  = (const int*)d_in[2];
  const float* apart = (const float*)d_in[3];
  const float* c1Wl  = (const float*)d_in[4];
  const float* c1Wr  = (const float*)d_in[5];
  const float* c1b   = (const float*)d_in[6];
  const float* c2Wl  = (const float*)d_in[7];
  const float* c2Wr  = (const float*)d_in[8];
  const float* c2b   = (const float*)d_in[9];
  const float* Wq    = (const float*)d_in[10];
  const float* bq    = (const float*)d_in[11];
  const float* Wk    = (const float*)d_in[12];
  const float* bk    = (const float*)d_in[13];
  const float* Wv    = (const float*)d_in[14];
  const float* bv    = (const float*)d_in[15];
  const float* f1W   = (const float*)d_in[16];
  const float* f1b   = (const float*)d_in[17];
  const float* f2W   = (const float*)d_in[18];
  const float* f2b   = (const float*)d_in[19];
  const float* f3W   = (const float*)d_in[20];
  const float* f3b   = (const float*)d_in[21];

  int E  = in_sizes[1] / 2;
  int N  = in_sizes[0] / (T_ * IN_);
  int Bn = in_sizes[2];
  size_t TNH = (size_t)T_ * N * H_;

  float* deg  = (float*)d_ws;
  float* dinv = deg + N;
  float* agg1 = dinv + N;
  float* bufA = agg1 + (size_t)T_ * N * IN_;
  float* bufB = bufA + TNH;
  float* bufC = bufB + TNH;
  float* femb = bufC + TNH;

  (void)hipMemsetAsync(deg, 0, (size_t)N * sizeof(float), stream);
  (void)hipMemsetAsync(agg1, 0, (size_t)T_ * N * IN_ * sizeof(float), stream);
  (void)hipMemsetAsync(bufC, 0, TNH * sizeof(float), stream);

  k_degree<<<(E + 255) / 256, 256, 0, stream>>>(ei, E, deg);
  k_deginv<<<(N + 255) / 256, 256, 0, stream>>>(deg, dinv, N);
  k_scatter12<<<dim3((E + 255) / 256, T_), 256, 0, stream>>>(ei, E, N, x, agg1);

  int ntiles = (N + 15) / 16;
  size_t sh1 = 2 * (128 * 40 * 2) + 128 * 4 + 8 * PW_C1;
  k_conv1<<<512, 256, sh1, stream>>>(x, agg1, dinv, c1Wl, c1Wr, c1b, bufA, N, ntiles);

  size_t shA = (size_t)3 * (128 * 136 * 2) + 3 * 128 * 4 + 8 * PW_ATTN;
  k_attn<<<768, 256, shA, stream>>>(bufA, bufB, nullptr, Wq, bq, Wk, bk, Wv, bv, N, 0);

  long long sc = (long long)E * 8;
  k_scatter128<<<dim3((unsigned)((sc + 255) / 256), T_), 256, 0, stream>>>(ei, E, N, bufB, bufC);

  size_t sh2 = 2 * (128 * 136 * 2) + 128 * 4 + 8 * PW_C2;
  k_conv2<<<512, 256, sh2, stream>>>(bufB, bufC, dinv, c2Wl, c2Wr, c2b, bufA, N, ntiles);

  k_attn<<<768, 256, shA, stream>>>(bufA, nullptr, femb, Wq, bq, Wk, bk, Wv, bv, N, 1);

  size_t shF = (size_t)256 * 168 * 2 + 128 * 264 * 2 + (256 + 128 + 128 + 4) * 4 + 4 * PW_FC;
  k_fc<<<dim3((Bn + 63) / 64), 128, shF, stream>>>(femb, nidx, apart,
                                                   f1W, f1b, f2W, f2b, f3W, f3b,
                                                   (float*)d_out, Bn);
}
